// VisionTransformer_23888608101370
// MI455X (gfx1250) — matrix operand order repaired and checked by reading, not compiled, not run on hardware
//
#include <hip/hip_runtime.h>
#include <stdint.h>

// ---------------- constants ----------------
constexpr int kBatch = 8;
constexpr int kD     = 256;
constexpr int kNH    = 8;
constexpr int kHD    = 32;          // head dim == one bf16 WMMA K-step
constexpr int kDepth = 6;
constexpr int kN     = 1025;        // tokens (1024 patches + cls)
constexpr int kM     = kBatch * kN; // 8200 rows in the residual stream
constexpr int kNPAD  = 1056;        // 33 * 32, padded key count
constexpr int kPD    = 192;         // patch dim 3*8*8
constexpr int kDh    = 1024;        // MLP hidden
constexpr int kNP    = 1024;        // patches per image
constexpr int kIMG   = 256;
constexpr float kEPS   = 1e-5f;
constexpr float kScale = 0.17677669529663687f; // 32^-0.5

typedef __attribute__((ext_vector_type(16))) __bf16 v16bf;
typedef __attribute__((ext_vector_type(8)))  float  v8f;

union FragA { v16bf v; uint32_t u[8]; };
union FragC { v8f   v; float    f[8]; };

__device__ __forceinline__ unsigned short f2bf(float x) {
  union { float f; uint32_t u; } cv; cv.f = x;
  const uint32_t u = cv.u;
  return (unsigned short)((u + 0x7FFFu + ((u >> 16) & 1u)) >> 16); // RNE
}

__device__ __forceinline__ v8f wmma_bf16(v16bf a, v16bf b, v8f c) {
  return __builtin_amdgcn_wmma_f32_16x16x32_bf16(false, a, false, b, (short)0, c,
                                                 false, false);
}

// A fragment (16x32 bf16). rowptr = &A[row_of_this_lane][k0].
// Layout: lanes 0-15 row M=lane, K=0..15 in pairs; lanes 16-31 same
// rows, K=16..31 (matches load_b16).
__device__ __forceinline__ v16bf load_a16(const unsigned short* rowptr, int lane) {
  FragA fa;
  const int ko = (lane < 16) ? 0 : 16;
#pragma unroll
  for (int p = 0; p < 8; ++p) {
    const int k = ko + 2 * p;
    fa.u[p] = *(const uint32_t*)(rowptr + k);
  }
  return fa.v;
}

// B fragment (32x16 bf16). Source "N-major": colptr = &W[col_of_lane][k0].
// Layout used here (contiguous k, same as load_a16): lanes 0-15 col N=lane, K=0..15 in halfword pairs; lanes 16-31
// col N=lane-16, K=16..31.
__device__ __forceinline__ v16bf load_b16(const unsigned short* colptr, int lane) {
  FragA fb;
  const int ko = (lane < 16) ? 0 : 16;
#pragma unroll
  for (int p = 0; p < 8; ++p)
    fb.u[p] = *(const uint32_t*)(colptr + ko + 2 * p);
  return fb.v;
}

// ---------------- generic bf16 WMMA GEMM ----------------
// One wave (32 thr) computes a 16x64 C tile. A: [Mrows,K] bf16 row-major.
// W: [Ncols][K] bf16 (transposed weights). Epilogues:
//   0 = out f32 = acc + bias
//   1 = out f32 = acc + bias + res   (residual add; res/out alias elementwise)
//   2 = outBf  = bf16(gelu(acc + bias))
//   3 = scatter acc+bias into q/k/v head-layout bf16 buffers
template <int EPI>
__global__ __launch_bounds__(32) void gemm_bf16_kernel(
    const unsigned short* __restrict__ A, const unsigned short* __restrict__ W,
    const float* __restrict__ bias, float* __restrict__ outF,
    unsigned short* __restrict__ outBf, const float* __restrict__ res,
    int Mrows, int Ncols, int K,
    unsigned short* __restrict__ qb, unsigned short* __restrict__ kb,
    unsigned short* __restrict__ vtb) {
  const int lane = threadIdx.x;
  const int tn = blockIdx.x;       // tile of 64 columns
  const int row0 = blockIdx.y * 16;
  const int nl = lane & 15;
  int m = row0 + nl;
  if (m >= Mrows) m = Mrows - 1;   // clamped duplicate loads; stores are guarded
  const unsigned short* arow = A + (size_t)m * K;
  const unsigned short* wrow = W + (size_t)(tn * 64 + nl) * K;
  v8f acc0 = {}, acc1 = {}, acc2 = {}, acc3 = {};
  for (int k0 = 0; k0 < K; k0 += 32) {
    if (k0 + 32 < K) {  // gfx1250 global_prefetch for the next K step
      __builtin_prefetch(arow + k0 + 32, 0, 3);
      __builtin_prefetch(wrow + k0 + 32, 0, 3);
    }
    const v16bf a = load_a16(arow + k0, lane);
    const unsigned short* wb = wrow + k0;
    const v16bf b0 = load_b16(wb, lane);
    const v16bf b1 = load_b16(wb + (size_t)16 * K, lane);
    const v16bf b2 = load_b16(wb + (size_t)32 * K, lane);
    const v16bf b3 = load_b16(wb + (size_t)48 * K, lane);
    acc0 = wmma_bf16(a, b0, acc0);
    acc1 = wmma_bf16(a, b1, acc1);
    acc2 = wmma_bf16(a, b2, acc2);
    acc3 = wmma_bf16(a, b3, acc3);
  }
  const int rb = (lane < 16) ? 0 : 8;   // f32 C layout: VGPR e -> row rb+e, col=nl
  FragC fr[4];
  fr[0].v = acc0; fr[1].v = acc1; fr[2].v = acc2; fr[3].v = acc3;
#pragma unroll
  for (int c = 0; c < 4; ++c) {
    const int col = tn * 64 + c * 16 + nl;
    const float bv = bias ? bias[col] : 0.0f;
#pragma unroll
    for (int e = 0; e < 8; ++e) {
      const int r = row0 + rb + e;
      if (r >= Mrows) continue;
      const float val = fr[c].f[e] + bv;
      if constexpr (EPI == 0) {
        outF[(size_t)r * Ncols + col] = val;
      } else if constexpr (EPI == 1) {
        outF[(size_t)r * Ncols + col] = val + res[(size_t)r * Ncols + col];
      } else if constexpr (EPI == 2) {
        const float g = 0.5f * val * (1.0f + erff(val * 0.70710678118654752f));
        outBf[(size_t)r * Ncols + col] = f2bf(g);
      } else {  // EPI == 3 : qkv scatter
        const int bb = r / kN, nn = r % kN;
        const int sel = col >> 8, w = col & 255, hh = w >> 5, dd = w & 31;
        const unsigned short bfv = f2bf(val);
        if (sel == 0)
          qb[((size_t)(bb * kNH + hh) * kNPAD + nn) * kHD + dd] = bfv;
        else if (sel == 1)
          kb[((size_t)(bb * kNH + hh) * kNPAD + nn) * kHD + dd] = bfv;
        else
          vtb[((size_t)(bb * kNH + hh) * kHD + dd) * kNPAD + nn] = bfv;
      }
    }
  }
}

// ---------------- flash attention ----------------
// One wave per (b, h, 16-q-row tile). K/V chunks (32 keys) are double-buffered
// in LDS via gfx1250 async global->LDS copies overlapped with compute.
// Softmax: tile-wide running max bound (5 ds_bpermute/chunk), row sums via a
// WMMA against an all-ones B matrix (C-layout rows match the O accumulator).
__global__ __launch_bounds__(32) void attn_kernel(
    const unsigned short* __restrict__ qbuf, const unsigned short* __restrict__ kbuf,
    const unsigned short* __restrict__ vtbuf, unsigned short* __restrict__ obf) {
  __shared__ __align__(16) unsigned short Kl[2][32 * 32]; // [key][dim]
  __shared__ __align__(16) unsigned short Vl[2][32 * 32]; // [dim][key]
  __shared__ __align__(16) unsigned short Plds[16 * 32];
  const int lane = threadIdx.x;
  const int bh = blockIdx.x;        // b*NH + h
  const int row0 = blockIdx.y * 16; // q tile
  const int bb = bh >> 3, hh = bh & 7;
  const unsigned short* q = qbuf + (size_t)bh * kNPAD * kHD;
  const uint64_t kbase = (uint64_t)(uintptr_t)(kbuf + (size_t)bh * kNPAD * kHD);
  const uint64_t vbase = (uint64_t)(uintptr_t)(vtbuf + (size_t)bh * kHD * kNPAD);
  const int nl = lane & 15;

  const v16bf aq = load_a16(q + (size_t)(row0 + nl) * kHD, lane); // padded rows = 0

  FragA ones;  // all-ones bf16 B fragment (1.0 = 0x3F80) for row sums
#pragma unroll
  for (int p = 0; p < 8; ++p) ones.u[p] = 0x3F803F80u;

  // Stage chunk c (keys c*32..c*32+31) into LDS buffer `buf`: 16 B per lane per
  // issue. K rows are 64 B contiguous; V rows are dim-major (even stride).
  auto stage = [&](int c, int buf) {
    const uint32_t kdst0 = (uint32_t)(uintptr_t)&Kl[buf][lane * 32];
    const uint32_t vdst0 = (uint32_t)(uintptr_t)&Vl[buf][lane * 32];
    const uint32_t ksrc0 = (uint32_t)((c * 32 + lane) * kHD * 2);   // bytes
    const uint32_t vsrc0 = (uint32_t)((lane * kNPAD + c * 32) * 2); // bytes
#pragma unroll
    for (int s = 0; s < 4; ++s) {
      asm volatile("global_load_async_to_lds_b128 %0, %1, %2"
                   :: "v"(kdst0 + s * 16), "v"(ksrc0 + s * 16), "s"(kbase)
                   : "memory");
      asm volatile("global_load_async_to_lds_b128 %0, %1, %2"
                   :: "v"(vdst0 + s * 16), "v"(vsrc0 + s * 16), "s"(vbase)
                   : "memory");
    }
  };

  FragC O0, O1;          // output dims 0-15 / 16-31, rows per C layout
  float lrow[8];
  float mcur = -1e30f;   // wave-uniform running max bound
#pragma unroll
  for (int e = 0; e < 8; ++e) { O0.f[e] = 0.f; O1.f[e] = 0.f; lrow[e] = 0.f; }

  stage(0, 0);
  constexpr int nchunks = kNPAD / 32; // 33
  for (int ci = 0; ci < nchunks; ++ci) {
    const int cur = ci & 1;
    asm volatile("s_wait_asynccnt 0" ::: "memory"); // chunk ci resident in LDS
    __syncthreads();
    if (ci + 1 < nchunks) stage(ci + 1, cur ^ 1);   // overlap next copy

    const int k0 = ci * 32;
    // S = q @ K^T  (B frag: lane = key, K pairs = head dims)
    const v16bf b0 = load_b16(&Kl[cur][(size_t)nl * 32], lane);
    const v16bf b1 = load_b16(&Kl[cur][(size_t)(16 + nl) * 32], lane);
    const v8f z = {};
    FragC s0, s1;
    s0.v = wmma_bf16(aq, b0, z);
    s1.v = wmma_bf16(aq, b1, z);
    const bool ok0 = (k0 + nl) < kN;
    const bool ok1 = (k0 + 16 + nl) < kN;
    float cm = -1e30f;
#pragma unroll
    for (int e = 0; e < 8; ++e) {
      const float a0 = ok0 ? s0.f[e] * kScale : -1e30f;
      const float a1 = ok1 ? s1.f[e] * kScale : -1e30f;
      s0.f[e] = a0; s1.f[e] = a1;
      cm = fmaxf(cm, fmaxf(a0, a1));
    }
#pragma unroll
    for (int off = 1; off < 32; off <<= 1)  // tile-wide max bound (wave-uniform)
      cm = fmaxf(cm, __shfl_xor(cm, off));
    const float mn = fmaxf(mcur, cm);
    const float corr = __expf(mcur - mn);
    mcur = mn;
#pragma unroll
    for (int e = 0; e < 8; ++e) { O0.f[e] *= corr; O1.f[e] *= corr; lrow[e] *= corr; }

    // P = exp(S - mn), masked; reshape C layout -> LDS [16 rows][32 keys]
    const int rb = (lane < 16) ? 0 : 8;
#pragma unroll
    for (int e = 0; e < 8; ++e) {
      const float p0 = ok0 ? __expf(s0.f[e] - mn) : 0.f;
      const float p1 = ok1 ? __expf(s1.f[e] - mn) : 0.f;
      Plds[(rb + e) * 32 + nl]      = f2bf(p0);
      Plds[(rb + e) * 32 + 16 + nl] = f2bf(p1);
    }
    __syncthreads();
    const v16bf pa = load_a16(Plds + nl * 32, lane);
    // row sums via WMMA with all-ones B (every column = rowsum)
    FragC ls; ls.v = wmma_bf16(pa, ones.v, z);
#pragma unroll
    for (int e = 0; e < 8; ++e) lrow[e] += ls.f[e];
    // O += P @ V  (B frag: lane = dim, K pairs = keys)
    const v16bf vb0 = load_b16(&Vl[cur][(size_t)nl * 32], lane);
    const v16bf vb1 = load_b16(&Vl[cur][(size_t)(16 + nl) * 32], lane);
    O0.v = wmma_bf16(pa, vb0, O0.v);
    O1.v = wmma_bf16(pa, vb1, O1.v);
    __syncthreads();
  }

  const int rb = (lane < 16) ? 0 : 8;
#pragma unroll
  for (int e = 0; e < 8; ++e) {
    const int r = row0 + rb + e;
    if (r < kN) {
      const float inv = 1.0f / lrow[e];
      const size_t base = ((size_t)bb * kN + r) * kD + hh * kHD;
      obf[base + nl]      = f2bf(O0.f[e] * inv);
      obf[base + 16 + nl] = f2bf(O1.f[e] * inv);
    }
  }
}

// ---------------- LayerNorm (one 256-thread block per row) ----------------
__global__ void ln_bf16_kernel(const float* __restrict__ x, const float* __restrict__ g,
                               const float* __restrict__ b, unsigned short* __restrict__ out) {
  __shared__ float s1[256];
  __shared__ float s2[256];
  const int r = blockIdx.x, c = threadIdx.x;
  const float v = x[(size_t)r * kD + c];
  s1[c] = v; s2[c] = v * v;
  __syncthreads();
  for (int o = 128; o > 0; o >>= 1) {
    if (c < o) { s1[c] += s1[c + o]; s2[c] += s2[c + o]; }
    __syncthreads();
  }
  const float mu  = s1[0] * (1.0f / kD);
  const float var = s2[0] * (1.0f / kD) - mu * mu;
  const float inv = rsqrtf(var + kEPS);
  out[(size_t)r * kD + c] = f2bf((v - mu) * inv * g[c] + b[c]);
}

__global__ void ln_cls_kernel(const float* __restrict__ t, const float* __restrict__ g,
                              const float* __restrict__ b, float* __restrict__ rep) {
  __shared__ float s1[256];
  __shared__ float s2[256];
  const int bb = blockIdx.x, c = threadIdx.x;
  const float v = t[(size_t)bb * kN * kD + c]; // cls row of batch bb
  s1[c] = v; s2[c] = v * v;
  __syncthreads();
  for (int o = 128; o > 0; o >>= 1) {
    if (c < o) { s1[c] += s1[c + o]; s2[c] += s2[c + o]; }
    __syncthreads();
  }
  const float mu  = s1[0] * (1.0f / kD);
  const float var = s2[0] * (1.0f / kD) - mu * mu;
  const float inv = rsqrtf(var + kEPS);
  rep[bb * kD + c] = (v - mu) * inv * g[c] + b[c];
}

// ---------------- small utility kernels ----------------
__global__ void wconv_kernel(const float* __restrict__ W, unsigned short* __restrict__ Wt,
                             int K, int Ncols, long inStride, long outStride) {
  const float* w = W + (size_t)blockIdx.z * inStride;
  unsigned short* o = Wt + (size_t)blockIdx.z * outStride;
  const int total = K * Ncols;
  for (int idx = blockIdx.x * blockDim.x + threadIdx.x; idx < total;
       idx += gridDim.x * blockDim.x) {
    const int k = idx / Ncols, n = idx % Ncols;
    o[(size_t)n * K + k] = f2bf(w[idx]);   // Wt[n][k] = W[k][n]
  }
}

__global__ void patch_kernel(const float* __restrict__ x, unsigned short* __restrict__ pbuf) {
  const int total = kBatch * kNP * kPD;
  for (int idx = blockIdx.x * blockDim.x + threadIdx.x; idx < total;
       idx += gridDim.x * blockDim.x) {
    const int col = idx % kPD;
    const int row = idx / kPD;
    const int bb = row >> 10, p = row & 1023;
    const int ph = p >> 5, pw = p & 31;
    const int c = col / 64, rem = col % 64;
    const int i = rem >> 3, j = rem & 7;
    pbuf[idx] = f2bf(x[(((size_t)bb * 3 + c) * kIMG + (ph * 8 + i)) * kIMG + (pw * 8 + j)]);
  }
}

__global__ void assemble_kernel(const float* __restrict__ tmb, const float* __restrict__ cls,
                                const float* __restrict__ pos, float* __restrict__ t) {
  const int total = kM * kD;
  for (int idx = blockIdx.x * blockDim.x + threadIdx.x; idx < total;
       idx += gridDim.x * blockDim.x) {
    const int c = idx & 255;
    const int rn = idx >> 8;
    const int bb = rn / kN, n = rn % kN;
    t[idx] = pos[n * kD + c] +
             ((n == 0) ? cls[c] : tmb[((size_t)bb * kNP + (n - 1)) * kD + c]);
  }
}

__global__ void heads_kernel(const float* __restrict__ rep, const float* __restrict__ wc,
                             const float* __restrict__ bc, const float* __restrict__ wd,
                             const float* __restrict__ bd, float* __restrict__ out) {
  const int tid = threadIdx.x;
  if (tid < 80) {
    const int bb = tid / 10, j = tid % 10;
    float s = bc[j];
    for (int i = 0; i < kD; ++i) s += rep[bb * kD + i] * wc[i * 10 + j];
    out[tid] = s;
  } else if (tid < 128) {
    const int t2 = tid - 80;
    const int bb = t2 / 6, j = t2 % 6;
    float s = bd[j];
    for (int i = 0; i < kD; ++i) s += rep[bb * kD + i] * wd[i * 6 + j];
    out[80 + t2] = s;
  }
}

// ---------------- host driver ----------------
extern "C" void kernel_launch(void* const* d_in, const int* in_sizes, int n_in,
                              void* d_out, int out_size, void* d_ws, size_t ws_size,
                              hipStream_t stream) {
  (void)in_sizes; (void)n_in; (void)out_size;
  const float* x       = (const float*)d_in[0];
  const float* proj_W  = (const float*)d_in[1];
  const float* proj_b  = (const float*)d_in[2];
  const float* cls_tok = (const float*)d_in[3];
  const float* pos_emb = (const float*)d_in[4];
  const float* qkv_W   = (const float*)d_in[5];
  const float* qkv_b   = (const float*)d_in[6];
  const float* ao_W    = (const float*)d_in[7];
  const float* ao_b    = (const float*)d_in[8];
  const float* ln1_g   = (const float*)d_in[9];
  const float* ln1_b   = (const float*)d_in[10];
  const float* ln2_g   = (const float*)d_in[11];
  const float* ln2_b   = (const float*)d_in[12];
  const float* m1_W    = (const float*)d_in[13];
  const float* m1_b    = (const float*)d_in[14];
  const float* m2_W    = (const float*)d_in[15];
  const float* m2_b    = (const float*)d_in[16];
  const float* norm_g  = (const float*)d_in[17];
  const float* norm_b  = (const float*)d_in[18];
  const float* fcW     = (const float*)d_in[19];
  const float* fcb     = (const float*)d_in[20];
  const float* fdW     = (const float*)d_in[21];
  const float* fdb     = (const float*)d_in[22];
  float* out = (float*)d_out;

  char* ws = (char*)d_ws;
  size_t off = 0;
  auto take = [&](size_t bytes) -> void* {
    off = (off + 255) & ~(size_t)255;
    void* p = ws + off;
    off += bytes;
    return p;
  };
  unsigned short* pb     = (unsigned short*)take((size_t)kBatch * kNP * kPD * 2);
  unsigned short* projWt = (unsigned short*)take((size_t)kD * kPD * 2);
  unsigned short* qkvWt  = (unsigned short*)take((size_t)kDepth * 3 * kD * kD * 2);
  unsigned short* aoWt   = (unsigned short*)take((size_t)kDepth * kD * kD * 2);
  unsigned short* m1Wt   = (unsigned short*)take((size_t)kDepth * kDh * kD * 2);
  unsigned short* m2Wt   = (unsigned short*)take((size_t)kDepth * kD * kDh * 2);
  float* tmb             = (float*)take((size_t)kBatch * kNP * kD * 4);
  float* t               = (float*)take((size_t)kM * kD * 4);
  unsigned short* hbf    = (unsigned short*)take((size_t)kM * kD * 2);
  unsigned short* qbuf   = (unsigned short*)take((size_t)kBatch * kNH * kNPAD * kHD * 2);
  unsigned short* kbuf   = (unsigned short*)take((size_t)kBatch * kNH * kNPAD * kHD * 2);
  unsigned short* vtbuf  = (unsigned short*)take((size_t)kBatch * kNH * kHD * kNPAD * 2);
  unsigned short* obf    = (unsigned short*)take((size_t)kM * kD * 2);
  unsigned short* hid    = (unsigned short*)take((size_t)kM * kDh * 2);
  float* rep             = (float*)take((size_t)kBatch * kD * 4);
  if (off > ws_size) return;

  // Weights -> bf16, transposed [N][K] for contiguous B-fragment loads.
  wconv_kernel<<<dim3(96, 1, 1), 256, 0, stream>>>(proj_W, projWt, kPD, kD, 0, 0);
  wconv_kernel<<<dim3(192, 1, kDepth), 256, 0, stream>>>(qkv_W, qkvWt, kD, 3 * kD,
      (long)kD * 3 * kD, (long)3 * kD * kD);
  wconv_kernel<<<dim3(96, 1, kDepth), 256, 0, stream>>>(ao_W, aoWt, kD, kD,
      (long)kD * kD, (long)kD * kD);
  wconv_kernel<<<dim3(192, 1, kDepth), 256, 0, stream>>>(m1_W, m1Wt, kD, kDh,
      (long)kD * kDh, (long)kDh * kD);
  wconv_kernel<<<dim3(192, 1, kDepth), 256, 0, stream>>>(m2_W, m2Wt, kDh, kD,
      (long)kDh * kD, (long)kD * kDh);

  // Zero q/k/v so padded rows (token 1025..1055) are exact zeros.
  hipMemsetAsync(qbuf, 0, (size_t)kBatch * kNH * kNPAD * kHD * 2, stream);
  hipMemsetAsync(kbuf, 0, (size_t)kBatch * kNH * kNPAD * kHD * 2, stream);
  hipMemsetAsync(vtbuf, 0, (size_t)kBatch * kNH * kHD * kNPAD * 2, stream);

  // Patch embed: extract -> WMMA GEMM -> assemble (cls + pos).
  patch_kernel<<<dim3(1024, 1, 1), 256, 0, stream>>>(x, pb);
  gemm_bf16_kernel<0><<<dim3(kD / 64, (kBatch * kNP) / 16), 32, 0, stream>>>(
      pb, projWt, proj_b, tmb, nullptr, nullptr, kBatch * kNP, kD, kPD,
      nullptr, nullptr, nullptr);
  assemble_kernel<<<dim3(1024, 1, 1), 256, 0, stream>>>(tmb, cls_tok, pos_emb, t);

  const int gy = (kM + 15) / 16; // 513 row tiles
  for (int l = 0; l < kDepth; ++l) {
    ln_bf16_kernel<<<dim3(kM, 1, 1), kD, 0, stream>>>(t, ln1_g + l * kD, ln1_b + l * kD, hbf);
    gemm_bf16_kernel<3><<<dim3(3 * kD / 64, gy), 32, 0, stream>>>(
        hbf, qkvWt + (size_t)l * 3 * kD * kD, qkv_b + l * 3 * kD,
        nullptr, nullptr, nullptr, kM, 3 * kD, kD, qbuf, kbuf, vtbuf);
    attn_kernel<<<dim3(kBatch * kNH, (kN + 15) / 16), 32, 0, stream>>>(qbuf, kbuf, vtbuf, obf);
    gemm_bf16_kernel<1><<<dim3(kD / 64, gy), 32, 0, stream>>>(
        obf, aoWt + (size_t)l * kD * kD, ao_b + l * kD, t, nullptr, t,
        kM, kD, kD, nullptr, nullptr, nullptr);
    ln_bf16_kernel<<<dim3(kM, 1, 1), kD, 0, stream>>>(t, ln2_g + l * kD, ln2_b + l * kD, hbf);
    gemm_bf16_kernel<2><<<dim3(kDh / 64, gy), 32, 0, stream>>>(
        hbf, m1Wt + (size_t)l * kDh * kD, m1_b + l * kDh, nullptr, hid, nullptr,
        kM, kDh, kD, nullptr, nullptr, nullptr);
    gemm_bf16_kernel<1><<<dim3(kD / 64, gy), 32, 0, stream>>>(
        hid, m2Wt + (size_t)l * kD * kDh, m2_b + l * kD, t, nullptr, t,
        kM, kD, kDh, nullptr, nullptr, nullptr);
  }

  ln_cls_kernel<<<dim3(kBatch, 1, 1), kD, 0, stream>>>(t, norm_g, norm_b, rep);
  heads_kernel<<<dim3(1, 1, 1), 128, 0, stream>>>(rep, fcW, fcb, fdW, fdb, out);
}
